// SelectiveSSM_35304631174086
// MI455X (gfx1250) — matrix operand order repaired and checked by reading, not compiled, not run on hardware
//
#include <hip/hip_runtime.h>
#include <math.h>

typedef _Float16 v16h __attribute__((ext_vector_type(16)));
typedef float    v8f  __attribute__((ext_vector_type(8)));

#define B_SZ    2
#define SEQ_L   1024
#define D_INNER 2048
#define D_STATE 16
#define DT_RANK 64
#define KTOT    (DT_RANK + 2 * D_STATE)   // 96
#define M_ROWS  (B_SZ * SEQ_L)            // 2048

// ---------------------------------------------------------------------------
// Async global->LDS path (gfx1250 GLOBAL_LOAD_ASYNC_TO_LDS_B128, ASYNCcnt).
// Guarded so the file compiles on toolchains without the builtin.
// Builtin signature (from hipcc diagnostics):
//   arg0: v4i*  (generic pointer to int vector_size(16))
//   arg1: __attribute__((address_space(3))) v4i*   (LDS pointer)
//   arg2, arg3: int (offset, cpol)
// ---------------------------------------------------------------------------
#if defined(__HIP_DEVICE_COMPILE__) && defined(__gfx1250__) && \
    __has_builtin(__builtin_amdgcn_global_load_async_to_lds_b128)
#define ASYNC_LDS 1
#else
#define ASYNC_LDS 0
#endif

#if ASYNC_LDS
typedef int v4i __attribute__((__vector_size__(4 * sizeof(int))));
#define GSRC(p) ((v4i*)(p))
#define LDST(p) ((__attribute__((address_space(3))) v4i*)(p))
__device__ __forceinline__ void async_wait0() {
#if __has_builtin(__builtin_amdgcn_s_wait_asynccnt)
    __builtin_amdgcn_s_wait_asynccnt(0);
#else
    asm volatile("s_wait_asynccnt 0" ::: "memory");
#endif
}
#endif

// ---------------------------------------------------------------------------
// WMMA helper: D = A(16x32 f16) * B(32x16 f16) + C(16x16 f32)
// ---------------------------------------------------------------------------
__device__ __forceinline__ v8f wmma_f16(v16h a, v16h b, v8f c) {
    return __builtin_amdgcn_wmma_f32_16x16x32_f16(
        /*neg_a=*/false, a, /*neg_b=*/false, b,
        /*c_mod=*/(short)0, c, /*reuse_a=*/false, /*reuse_b=*/false);
}

// Fragment layouts per ISA 7.12.2:
//  A (16-bit, 16x32): K(e,lane) = baseKA + e + (e>=8 ? 8 : 0), baseKA = (lane<16)?0:8
//  B (16-bit, 32x16): K(e,lane) = baseKA + e + (e>=8 ? 8 : 0), same order as A
//  C/D (f32, 16x16):  VGPR v, lane l -> M = v + (l<16?0:8), N = l&15

// GEMM 1: xz[m, k] = sum_d x[m, d] * W_x[k, d]    (M=2048, N=96, K=2048)
__global__ void __launch_bounds__(128)
gemm_xz_kernel(const float* __restrict__ x, const float* __restrict__ Wx,
               float* __restrict__ xz) {
    const int wave    = (blockIdx.x * blockDim.x + threadIdx.x) >> 5;
    const int lane    = threadIdx.x & 31;
    const int nTilesN = KTOT / 16;                 // 6
    const int mt = wave / nTilesN, nt = wave % nTilesN;
    const int m0 = mt * 16, n0 = nt * 16;
    const int hiHalf = lane >> 4;
    const int l15    = lane & 15;
    const int aRow   = m0 + l15;
    const int bCol   = n0 + l15;
    const int baseKA = hiHalf * 8;
    const int baseKB = hiHalf * 16;

    v8f acc = {0.f, 0.f, 0.f, 0.f, 0.f, 0.f, 0.f, 0.f};

    for (int k0 = 0; k0 < D_INNER; k0 += 32) {
        v16h ah, al, bh, bl;
#pragma unroll
        for (int e = 0; e < 16; ++e) {
            const int ka = k0 + baseKA + e + ((e & 8) ? 8 : 0);
            const float av = x[(size_t)aRow * D_INNER + ka];
            const _Float16 avh = (_Float16)av;
            ah[e] = avh;
            al[e] = (_Float16)(av - (float)avh);

            const int kb = k0 + baseKA + e + ((e & 8) ? 8 : 0);
            const float bv = Wx[(size_t)bCol * D_INNER + kb];
            const _Float16 bvh = (_Float16)bv;
            bh[e] = bvh;
            bl[e] = (_Float16)(bv - (float)bvh);
        }
        acc = wmma_f16(ah, bh, acc);   // hi*hi
        acc = wmma_f16(al, bh, acc);   // lo*hi
        acc = wmma_f16(ah, bl, acc);   // hi*lo  (~f32 accuracy)
    }

#pragma unroll
    for (int v = 0; v < 8; ++v) {
        const int m = m0 + v + hiHalf * 8;
        xz[(size_t)m * KTOT + n0 + l15] = acc[v];
    }
}

// GEMM 2 (+bias+softplus): delta[m, d] = softplus(sum_r xz[m, r] * W_dt[d, r] + b_dt[d])
// (M=2048, N=2048, K=64)
__global__ void __launch_bounds__(256)
gemm_delta_kernel(const float* __restrict__ xz, const float* __restrict__ Wdt,
                  const float* __restrict__ bdt, float* __restrict__ delta) {
    const int wave    = (blockIdx.x * blockDim.x + threadIdx.x) >> 5;
    const int lane    = threadIdx.x & 31;
    const int nTilesN = D_INNER / 16;              // 128
    const int mt = wave / nTilesN, nt = wave % nTilesN;
    const int m0 = mt * 16, n0 = nt * 16;
    const int hiHalf = lane >> 4;
    const int l15    = lane & 15;
    const int aRow   = m0 + l15;
    const int dCol   = n0 + l15;
    const int baseKA = hiHalf * 8;
    const int baseKB = hiHalf * 16;

    v8f acc = {0.f, 0.f, 0.f, 0.f, 0.f, 0.f, 0.f, 0.f};

#pragma unroll
    for (int k0 = 0; k0 < DT_RANK; k0 += 32) {
        v16h ah, al, bh, bl;
#pragma unroll
        for (int e = 0; e < 16; ++e) {
            const int ka = k0 + baseKA + e + ((e & 8) ? 8 : 0);
            const float av = xz[(size_t)aRow * KTOT + ka];
            const _Float16 avh = (_Float16)av;
            ah[e] = avh;
            al[e] = (_Float16)(av - (float)avh);

            const int kb = k0 + baseKA + e + ((e & 8) ? 8 : 0);
            const float bv = Wdt[(size_t)dCol * DT_RANK + kb];
            const _Float16 bvh = (_Float16)bv;
            bh[e] = bvh;
            bl[e] = (_Float16)(bv - (float)bvh);
        }
        acc = wmma_f16(ah, bh, acc);
        acc = wmma_f16(al, bh, acc);
        acc = wmma_f16(ah, bl, acc);
    }

    const float bias = bdt[dCol];
#pragma unroll
    for (int v = 0; v < 8; ++v) {
        const int m = m0 + v + hiHalf * 8;
        const float t = acc[v] + bias;
        const float sp = (t > 20.f) ? t : log1pf(__expf(t));  // softplus
        delta[(size_t)m * D_INNER + dCol] = sp;
    }
}

// ---------------------------------------------------------------------------
// Selective scan. 4 lanes per channel d (4 states each), LDS-tiled over time.
// Staging uses GLOBAL_LOAD_ASYNC_TO_LDS_B128 when available (ASYNCcnt path),
// which bypasses VGPRs entirely; falls back to plain loads otherwise.
// ---------------------------------------------------------------------------
#define TT   64    // timesteps per LDS tile
#define DBLK 64    // channels per block

__global__ void __launch_bounds__(256)
scan_kernel(const float* __restrict__ x, const float* __restrict__ xz,
            const float* __restrict__ delta, const float* __restrict__ A_log,
            const float* __restrict__ Dp, float* __restrict__ out) {
    __shared__ float sBC[TT][32];      // cols 0..15 = B_in, 16..31 = C_in
    __shared__ float sDelta[TT][DBLK];
    __shared__ float sX[TT][DBLK];

    const int tid = threadIdx.x;
    const int b   = blockIdx.y;
    const int d0  = blockIdx.x * DBLK;
    const int dl  = tid >> 2;          // 0..63 : channel within block
    const int grp = tid & 3;           // 0..3  : state group
    const int d   = d0 + dl;
    const int n0  = grp * 4;

    float A4[4], h4[4] = {0.f, 0.f, 0.f, 0.f};
#pragma unroll
    for (int j = 0; j < 4; ++j)
        A4[j] = -__expf(A_log[(size_t)d * D_STATE + n0 + j]);
    const float Dd = Dp[d];

    for (int t0 = 0; t0 < SEQ_L; t0 += TT) {
        __syncthreads();   // previous tile's consumers done before overwrite

#if ASYNC_LDS
        // --- B/C slice: TT*32 floats = 512 float4, 2 per thread ---
#pragma unroll
        for (int i = 0; i < (TT * 32) / (256 * 4); ++i) {
            const int idx = tid + i * 256;              // float4 index
            const int tt = idx >> 3, c4 = idx & 7;      // 8 float4 per row
            __builtin_amdgcn_global_load_async_to_lds_b128(
                GSRC(&xz[(size_t)(b * SEQ_L + t0 + tt) * KTOT + DT_RANK + c4 * 4]),
                LDST(&sBC[tt][c4 * 4]), 0, 0);
        }
        // --- delta & x slices: TT*DBLK floats = 1024 float4 each, 4/thread ---
#pragma unroll
        for (int i = 0; i < (TT * DBLK) / (256 * 4); ++i) {
            const int idx = tid + i * 256;
            const int tt = idx >> 4, c4 = idx & 15;     // 16 float4 per row
            __builtin_amdgcn_global_load_async_to_lds_b128(
                GSRC(&delta[(size_t)(b * SEQ_L + t0 + tt) * D_INNER + d0 + c4 * 4]),
                LDST(&sDelta[tt][c4 * 4]), 0, 0);
            __builtin_amdgcn_global_load_async_to_lds_b128(
                GSRC(&x[(size_t)(b * SEQ_L + t0 + tt) * D_INNER + d0 + c4 * 4]),
                LDST(&sX[tt][c4 * 4]), 0, 0);
        }
        async_wait0();
#else
        // Fallback: stage through VGPRs
#pragma unroll
        for (int i = 0; i < (TT * 32) / 256; ++i) {
            const int idx = tid + i * 256;
            const int tt = idx >> 5, c = idx & 31;
            sBC[tt][c] = xz[(size_t)(b * SEQ_L + t0 + tt) * KTOT + DT_RANK + c];
        }
#pragma unroll
        for (int i = 0; i < (TT * DBLK) / 256; ++i) {
            const int idx = tid + i * 256;
            const int tt = idx >> 6, c = idx & 63;
            sDelta[tt][c] = delta[(size_t)(b * SEQ_L + t0 + tt) * D_INNER + d0 + c];
            sX[tt][c]     = x[(size_t)(b * SEQ_L + t0 + tt) * D_INNER + d0 + c];
        }
#endif
        // Hint the prefetcher at the next time-tile of delta
        if (t0 + TT < SEQ_L)
            __builtin_prefetch(&delta[(size_t)(b * SEQ_L + t0 + TT) * D_INNER + d0], 0, 1);
        __syncthreads();

        for (int tt = 0; tt < TT; ++tt) {
            const float dt = sDelta[tt][dl];
            const float xv = sX[tt][dl];
            const float dx = dt * xv;
            float y = 0.f;
#pragma unroll
            for (int j = 0; j < 4; ++j) {
                const float ab = __expf(dt * A4[j]);           // A_bar
                h4[j] = fmaf(ab, h4[j], dx * sBC[tt][n0 + j]); // h = A h + dBx
                y = fmaf(h4[j], sBC[tt][16 + n0 + j], y);      // y += h*C
            }
            // reduce partial y across the 4 lanes owning this channel
            y += __shfl_xor(y, 1);
            y += __shfl_xor(y, 2);
            if (grp == 0) {
                // y is written once, never re-read: keep it out of L2's way
                __builtin_nontemporal_store(
                    fmaf(Dd, xv, y),
                    &out[(size_t)(b * SEQ_L + t0 + tt) * D_INNER + d]);
            }
        }
    }
}

// ---------------------------------------------------------------------------
extern "C" void kernel_launch(void* const* d_in, const int* in_sizes, int n_in,
                              void* d_out, int out_size, void* d_ws, size_t ws_size,
                              hipStream_t stream) {
    const float* x     = (const float*)d_in[0];
    const float* Wx    = (const float*)d_in[1];
    const float* Wdt   = (const float*)d_in[2];
    const float* bdt   = (const float*)d_in[3];
    const float* A_log = (const float*)d_in[4];
    const float* Dp    = (const float*)d_in[5];
    float* out = (float*)d_out;

    // Workspace: delta (B*L*D floats = 16MB) then xz (M*96 floats = 0.75MB)
    float* deltaBuf = (float*)d_ws;
    float* xzBuf    = deltaBuf + (size_t)B_SZ * SEQ_L * D_INNER;

    // GEMM1: (2048/16)*(96/16) = 768 wave-tiles, 4 waves/block
    gemm_xz_kernel<<<768 / 4, 128, 0, stream>>>(x, Wx, xzBuf);
    // GEMM2: (2048/16)*(2048/16) = 16384 wave-tiles, 8 waves/block
    gemm_delta_kernel<<<16384 / 8, 256, 0, stream>>>(xzBuf, Wdt, bdt, deltaBuf);
    // Scan: 32 channel-blocks x 2 batches, 256 threads
    scan_kernel<<<dim3(D_INNER / DBLK, B_SZ), 256, 0, stream>>>(
        x, xzBuf, deltaBuf, A_log, Dp, out);
}
